// ApneaDetectionModel_68977174773955
// MI455X (gfx1250) — matrix operand order repaired and checked by reading, not compiled, not run on hardware
//
#include <hip/hip_runtime.h>
#include <hip/hip_bf16.h>

typedef __attribute__((ext_vector_type(16))) _Float16 v16h;
typedef __attribute__((ext_vector_type(8)))  float    v8f;
typedef _Float16 half_t;

#define D128 128
#define NBAND 5
#define NCHAN 3
#define BATCH 128
#define NTOK  15          // NBAND*NCHAN
#define NSEQ  1920        // BATCH*NTOK
#define T0 3840
#define T1 1920
#define T2 960
#define T3 480
#define C1 32
#define C2 64
#define C3 128

__device__ __forceinline__ float gelu_f(float x) {
    return 0.5f * x * (1.0f + erff(x * 0.70710678118654752f));
}

__device__ __forceinline__ float block_reduce_sum(float v, float* red, int tid) {
    red[tid] = v; __syncthreads();
    #pragma unroll
    for (int s = 64; s > 0; s >>= 1) {
        if (tid < s) red[tid] += red[tid + s];
        __syncthreads();
    }
    float r = red[0]; __syncthreads();
    return r;
}

__device__ __forceinline__ float layernorm_val(float x, const float* g, const float* b,
                                               float* red, int tid) {
    float mean = block_reduce_sum(x, red, tid) * (1.0f / 128.0f);
    float xm = x - mean;
    float var = block_reduce_sum(xm * xm, red, tid) * (1.0f / 128.0f);
    return xm * rsqrtf(var + 1e-5f) * g[tid] + b[tid];
}

// ---------------------------------------------------------------------------
// Fold BN (scale/shift) into conv weights, convert to f16 im2col layout.
// W2h[o][k], k = tap*32 + cin ; W3h[o][k], k = tap*64 + cin.
// ---------------------------------------------------------------------------
__global__ void prep_kernel(const float* c1w, const float* g1, const float* b1, const float* m1, const float* v1,
                            const float* c2w, const float* g2, const float* b2, const float* m2, const float* v2,
                            const float* c3w, const float* g3, const float* b3, const float* m3, const float* v3,
                            float* W1f, float* B1, half_t* W2h, float* B2, half_t* W3h, float* B3) {
    int i = blockIdx.x * blockDim.x + threadIdx.x;
    if (i < C1 * 9) {
        int o = i / 9;
        float s = g1[o] * rsqrtf(v1[o] + 1e-5f);
        W1f[i] = c1w[i] * s;
    }
    if (i < C1) { float s = g1[i] * rsqrtf(v1[i] + 1e-5f); B1[i] = b1[i] - m1[i] * s; }
    if (i < C2 * 224) {
        int o = i / 224, r = i % 224, tap = r / 32, ci = r % 32;
        float s = g2[o] * rsqrtf(v2[o] + 1e-5f);
        W2h[i] = (half_t)(c2w[o * 224 + ci * 7 + tap] * s);
    }
    if (i < C2) { float s = g2[i] * rsqrtf(v2[i] + 1e-5f); B2[i] = b2[i] - m2[i] * s; }
    if (i < C3 * 320) {
        int o = i / 320, r = i % 320, tap = r / 64, ci = r % 64;
        float s = g3[o] * rsqrtf(v3[o] + 1e-5f);
        W3h[i] = (half_t)(c3w[o * 320 + ci * 5 + tap] * s);
    }
    if (i < C3) { float s = g3[i] * rsqrtf(v3[i] + 1e-5f); B3[i] = b3[i] - m3[i] * s; }
}

// ---------------------------------------------------------------------------
// conv1: 1 -> 32 ch, k=9, stride 2, pad 4, + BN + gelu.  Memory bound, scalar.
// One block per sequence; input row staged in LDS.
// ---------------------------------------------------------------------------
__global__ void __launch_bounds__(256) conv1_kernel(const float* __restrict__ x,
                                                    const float* __restrict__ W1f,
                                                    const float* __restrict__ B1,
                                                    half_t* __restrict__ y1) {
    __shared__ float xs[T0];
    __shared__ float ws[C1 * 9];
    __shared__ float bs[C1];
    int seq = blockIdx.x, tid = threadIdx.x;
    const float* xrow = x + (size_t)seq * T0;
    for (int i = tid; i < T0; i += 256) xs[i] = xrow[i];
    for (int i = tid; i < C1 * 9; i += 256) ws[i] = W1f[i];
    if (tid < C1) bs[tid] = B1[tid];
    __syncthreads();
    for (int idx = tid; idx < C1 * T1; idx += 256) {
        int o = idx / T1, t = idx % T1;
        float acc = bs[o];
        int base = 2 * t - 4;
        #pragma unroll
        for (int j = 0; j < 9; ++j) {
            int xi = base + j;
            float xv = (xi >= 0 && xi < T0) ? xs[xi] : 0.0f;
            acc += xv * ws[o * 9 + j];
        }
        y1[((size_t)seq * C1 + o) * T1 + t] = (half_t)gelu_f(acc);
    }
}

// ---------------------------------------------------------------------------
// conv2 as WMMA GEMM: M=64 out-ch, K=224 (7 taps x 32 cin), N=64 positions per
// block.  LDS window stored x-major so B fragments are contiguous 32B chunks.
// 4 waves; each owns a 16-channel slab and 4 N-tiles (7*4 = 28 WMMA / wave).
// ---------------------------------------------------------------------------
__global__ void __launch_bounds__(128) conv2_wmma_kernel(const half_t* __restrict__ y1,
                                                         const half_t* __restrict__ W2h,
                                                         const float* __restrict__ B2,
                                                         half_t* __restrict__ y2) {
    __shared__ half_t lds[136 * C1];                    // [x][cin]
    int seq = blockIdx.y, t0 = blockIdx.x * 64, tid = threadIdx.x;
    int xbase = 2 * t0 - 3;
    for (int i = tid; i < 136 * C1; i += 128) {
        int xi = i >> 5, c = i & 31;
        int xg = xbase + xi;
        half_t v = (half_t)0.0f;
        if (xg >= 0 && xg < T1) v = y1[((size_t)seq * C1 + c) * T1 + xg];
        lds[i] = v;
    }
    __syncthreads();
    int lane = tid & 31, wave = tid >> 5;
    int half_ = lane >> 4, row = lane & 15;
    int mbase = wave * 16;
    v8f acc[4] = {};
    #pragma unroll
    for (int s = 0; s < 7; ++s) {
        v16h a;
        const half_t* wp = W2h + (mbase + row) * 224 + s * 32 + half_ * 16;
        #pragma unroll
        for (int e = 0; e < 8; ++e) { a[e] = wp[e]; a[e + 8] = wp[8 + e]; }
        #pragma unroll
        for (int nt = 0; nt < 4; ++nt) {
            int ncol = nt * 16 + row;
            const half_t* bp = &lds[(2 * ncol + s) * C1 + half_ * 16];
            v16h bf;
            #pragma unroll
            for (int e = 0; e < 16; ++e) bf[e] = bp[e];
            acc[nt] = __builtin_amdgcn_wmma_f32_16x16x32_f16(
                false, a, false, bf, (short)0, acc[nt], false, false);
        }
    }
    #pragma unroll
    for (int nt = 0; nt < 4; ++nt) {
        int t = t0 + nt * 16 + row;
        #pragma unroll
        for (int v = 0; v < 8; ++v) {
            int m = mbase + v + 8 * half_;
            float val = gelu_f(acc[nt][v] + B2[m]);
            y2[((size_t)seq * C2 + m) * T2 + t] = (half_t)val;
        }
    }
}

// ---------------------------------------------------------------------------
// conv3 as WMMA GEMM: M=128 out-ch, K=320 (5 taps x 64 cin), N=32 positions.
// 8 waves; each owns a 16-channel slab and 2 N-tiles (10*2 = 20 WMMA / wave).
// ---------------------------------------------------------------------------
__global__ void __launch_bounds__(256) conv3_wmma_kernel(const half_t* __restrict__ y2,
                                                         const half_t* __restrict__ W3h,
                                                         const float* __restrict__ B3,
                                                         half_t* __restrict__ feat) {
    __shared__ half_t lds[68 * C2];                     // [x][cin]
    int seq = blockIdx.y, t0 = blockIdx.x * 32, tid = threadIdx.x;
    int xbase = 2 * t0 - 2;
    for (int i = tid; i < 68 * C2; i += 256) {
        int xi = i >> 6, c = i & 63;
        int xg = xbase + xi;
        half_t v = (half_t)0.0f;
        if (xg >= 0 && xg < T2) v = y2[((size_t)seq * C2 + c) * T2 + xg];
        lds[i] = v;
    }
    __syncthreads();
    int lane = tid & 31, wave = tid >> 5;
    int half_ = lane >> 4, row = lane & 15;
    int mbase = wave * 16;
    v8f acc[2] = {};
    #pragma unroll
    for (int s = 0; s < 10; ++s) {
        int tap = s >> 1, cinb = (s & 1) * 32;
        v16h a;
        const half_t* wp = W3h + (mbase + row) * 320 + s * 32 + half_ * 16;
        #pragma unroll
        for (int e = 0; e < 8; ++e) { a[e] = wp[e]; a[e + 8] = wp[8 + e]; }
        #pragma unroll
        for (int nt = 0; nt < 2; ++nt) {
            int ncol = nt * 16 + row;
            const half_t* bp = &lds[(2 * ncol + tap) * C2 + cinb + half_ * 16];
            v16h bf;
            #pragma unroll
            for (int e = 0; e < 16; ++e) bf[e] = bp[e];
            acc[nt] = __builtin_amdgcn_wmma_f32_16x16x32_f16(
                false, a, false, bf, (short)0, acc[nt], false, false);
        }
    }
    #pragma unroll
    for (int nt = 0; nt < 2; ++nt) {
        int t = t0 + nt * 16 + row;
        #pragma unroll
        for (int v = 0; v < 8; ++v) {
            int m = mbase + v + 8 * half_;
            float val = gelu_f(acc[nt][v] + B3[m]);
            feat[((size_t)seq * C3 + m) * T3 + t] = (half_t)val;
        }
    }
}

// ---------------------------------------------------------------------------
// AttnPool1D over T=480 + band/chan embeddings.  One block per sequence.
// ---------------------------------------------------------------------------
__global__ void __launch_bounds__(128) attnpool_kernel(const half_t* __restrict__ feat,
                                                       const float* __restrict__ pool_w,
                                                       const float* __restrict__ pool_b,
                                                       const float* __restrict__ band_emb,
                                                       const float* __restrict__ chan_emb,
                                                       float* __restrict__ tok) {
    __shared__ float sc[T3];
    __shared__ float red[128];
    __shared__ float pw[D128];
    int seq = blockIdx.x, tid = threadIdx.x;
    const half_t* F = feat + (size_t)seq * D128 * T3;
    pw[tid] = pool_w[tid];
    __syncthreads();
    for (int t = tid; t < T3; t += 128) {
        float acc = 0.0f;
        for (int d = 0; d < D128; ++d) acc += (float)F[d * T3 + t] * pw[d];
        sc[t] = acc + pool_b[0];
    }
    __syncthreads();
    float mx = -1e30f;
    for (int t = tid; t < T3; t += 128) mx = fmaxf(mx, sc[t]);
    red[tid] = mx; __syncthreads();
    #pragma unroll
    for (int s = 64; s > 0; s >>= 1) {
        if (tid < s) red[tid] = fmaxf(red[tid], red[tid + s]);
        __syncthreads();
    }
    float M = red[0]; __syncthreads();
    float ls = 0.0f;
    for (int t = tid; t < T3; t += 128) { float e = __expf(sc[t] - M); sc[t] = e; ls += e; }
    float S = block_reduce_sum(ls, red, tid);
    int d = tid;
    float acc = 0.0f;
    for (int t = 0; t < T3; ++t) acc += (float)F[d * T3 + t] * sc[t];
    acc /= S;
    int token = seq % NTOK;
    int band = token / NCHAN, chan = token % NCHAN;
    tok[(size_t)seq * D128 + d] = acc + band_emb[band * D128 + d] + chan_emb[chan * D128 + d];
}

// ---------------------------------------------------------------------------
// Attention block 0: per (batch, channel): q = stage token, 5 kv = band tokens.
// MHA(H=4) + LN0 + FFN(128->512->128) + LN0.  One 128-thread block per row.
// ---------------------------------------------------------------------------
__global__ void __launch_bounds__(128) attn0_kernel(const int* __restrict__ stage,
                                                    const float* __restrict__ tok,
                                                    const float* __restrict__ stage_token,
                                                    const float* __restrict__ stage_emb,
                                                    const float* __restrict__ in_w, const float* __restrict__ in_b,
                                                    const float* __restrict__ out_w, const float* __restrict__ out_b,
                                                    const float* __restrict__ ln0_g, const float* __restrict__ ln0_b,
                                                    const float* __restrict__ w1, const float* __restrict__ b1,
                                                    const float* __restrict__ w2, const float* __restrict__ b2,
                                                    float* __restrict__ chan_sum) {
    __shared__ float sQin[D128], sQp[D128];
    __shared__ float sKV[NBAND][D128], sKp[NBAND][D128], sVp[NBAND][D128];
    __shared__ float sO[D128], sYb[D128], sH1[512], sS[NBAND * 4];
    __shared__ float red[128];
    int bc = blockIdx.x, b = bc / NCHAN, c = bc % NCHAN;
    int tid = threadIdx.x, d = tid;
    int st = stage[b];
    float qin = stage_token[d] + stage_emb[st * D128 + d];
    sQin[d] = qin;
    for (int j = 0; j < NBAND; ++j)
        sKV[j][d] = tok[(size_t)(b * NTOK + j * NCHAN + c) * D128 + d];
    __syncthreads();
    float aq = in_b[d];
    {
        const float* wq = in_w + (size_t)d * D128;
        for (int e = 0; e < D128; ++e) aq += sQin[e] * wq[e];
    }
    sQp[d] = aq;
    for (int j = 0; j < NBAND; ++j) {
        float ak = in_b[D128 + d], av = in_b[2 * D128 + d];
        const float* wk = in_w + (size_t)(D128 + d) * D128;
        const float* wv = in_w + (size_t)(2 * D128 + d) * D128;
        for (int e = 0; e < D128; ++e) { float kv = sKV[j][e]; ak += kv * wk[e]; av += kv * wv[e]; }
        sKp[j][d] = ak; sVp[j][d] = av;
    }
    __syncthreads();
    if (tid < NBAND * 4) {
        int j = tid >> 2, h = tid & 3;
        float s = 0.0f;
        for (int dd = 0; dd < 32; ++dd) s += sQp[h * 32 + dd] * sKp[j][h * 32 + dd];
        sS[h * NBAND + j] = s * 0.17677669529663687f;
    }
    __syncthreads();
    if (tid < 4) {
        int h = tid;
        float m = -1e30f;
        for (int j = 0; j < NBAND; ++j) m = fmaxf(m, sS[h * NBAND + j]);
        float sum = 0.0f;
        for (int j = 0; j < NBAND; ++j) { float e = __expf(sS[h * NBAND + j] - m); sS[h * NBAND + j] = e; sum += e; }
        for (int j = 0; j < NBAND; ++j) sS[h * NBAND + j] /= sum;
    }
    __syncthreads();
    {
        int h = d >> 5;
        float o = 0.0f;
        for (int j = 0; j < NBAND; ++j) o += sS[h * NBAND + j] * sVp[j][d];
        sO[d] = o;
    }
    __syncthreads();
    float ao = out_b[d];
    {
        const float* wr = out_w + (size_t)d * D128;
        for (int e = 0; e < D128; ++e) ao += sO[e] * wr[e];
    }
    float x = ao + qin;
    float yb = layernorm_val(x, ln0_g, ln0_b, red, tid);
    sYb[d] = yb; __syncthreads();
    for (int f = tid; f < 512; f += 128) {
        float acc = b1[f];
        const float* wr = w1 + (size_t)f * D128;
        for (int e = 0; e < D128; ++e) acc += sYb[e] * wr[e];
        sH1[f] = gelu_f(acc);
    }
    __syncthreads();
    float ff = b2[d];
    {
        const float* wr = w2 + (size_t)d * 512;
        for (int f = 0; f < 512; ++f) ff += sH1[f] * wr[f];
    }
    float out = layernorm_val(yb + ff, ln0_g, ln0_b, red, tid);
    chan_sum[(size_t)bc * D128 + d] = out;
}

// ---------------------------------------------------------------------------
// Attention block 1 + head: self-attn over 3 channel tokens, LN1, FFN, LN2,
// mean over channels, MLP head -> logit.  One 128-thread block per batch item.
// ---------------------------------------------------------------------------
__global__ void __launch_bounds__(128) attn1_kernel(const float* __restrict__ cs,
                                                    const float* __restrict__ in_w, const float* __restrict__ in_b,
                                                    const float* __restrict__ out_w, const float* __restrict__ out_b,
                                                    const float* __restrict__ ln1_g, const float* __restrict__ ln1_b,
                                                    const float* __restrict__ ln2_g, const float* __restrict__ ln2_b,
                                                    const float* __restrict__ w1, const float* __restrict__ b1,
                                                    const float* __restrict__ w2, const float* __restrict__ b2,
                                                    const float* __restrict__ h_w1, const float* __restrict__ h_b1,
                                                    const float* __restrict__ h_w2, const float* __restrict__ h_b2,
                                                    float* __restrict__ outp) {
    __shared__ float sX[NCHAN][D128], sQp[NCHAN][D128], sKp[NCHAN][D128], sVp[NCHAN][D128];
    __shared__ float sO[NCHAN][D128], sZ[NCHAN][D128], sZ2[NCHAN][D128];
    __shared__ float sH1[NCHAN * 512];
    __shared__ float sS[NCHAN * 4 * NCHAN];
    __shared__ float sZm[D128];
    __shared__ float red[128];
    int b = blockIdx.x, tid = threadIdx.x, d = tid;
    for (int c = 0; c < NCHAN; ++c) sX[c][d] = cs[(size_t)(b * NCHAN + c) * D128 + d];
    __syncthreads();
    for (int c = 0; c < NCHAN; ++c) {
        float aq = in_b[d], ak = in_b[D128 + d], av = in_b[2 * D128 + d];
        const float* wq = in_w + (size_t)d * D128;
        const float* wk = in_w + (size_t)(D128 + d) * D128;
        const float* wv = in_w + (size_t)(2 * D128 + d) * D128;
        for (int e = 0; e < D128; ++e) {
            float xv = sX[c][e];
            aq += xv * wq[e]; ak += xv * wk[e]; av += xv * wv[e];
        }
        sQp[c][d] = aq; sKp[c][d] = ak; sVp[c][d] = av;
    }
    __syncthreads();
    if (tid < NCHAN * 4 * NCHAN) {                 // 36 (cq,h,ck) score slots
        int cq = tid / 12, h = (tid / 3) & 3, ck = tid % 3;
        float s = 0.0f;
        for (int dd = 0; dd < 32; ++dd) s += sQp[cq][h * 32 + dd] * sKp[ck][h * 32 + dd];
        sS[(cq * 4 + h) * NCHAN + ck] = s * 0.17677669529663687f;
    }
    __syncthreads();
    if (tid < NCHAN * 4) {
        int cq = tid >> 2, h = tid & 3;
        float* p = &sS[(cq * 4 + h) * NCHAN];
        float m = fmaxf(p[0], fmaxf(p[1], p[2]));
        float e0 = __expf(p[0] - m), e1 = __expf(p[1] - m), e2 = __expf(p[2] - m);
        float s = e0 + e1 + e2;
        p[0] = e0 / s; p[1] = e1 / s; p[2] = e2 / s;
    }
    __syncthreads();
    {
        int h = d >> 5;
        for (int c = 0; c < NCHAN; ++c) {
            float o = 0.0f;
            for (int ck = 0; ck < NCHAN; ++ck) o += sS[(c * 4 + h) * NCHAN + ck] * sVp[ck][d];
            sO[c][d] = o;
        }
    }
    __syncthreads();
    for (int c = 0; c < NCHAN; ++c) {
        float ao = out_b[d];
        const float* wr = out_w + (size_t)d * D128;
        for (int e = 0; e < D128; ++e) ao += sO[c][e] * wr[e];
        sZ[c][d] = layernorm_val(ao + sX[c][d], ln1_g, ln1_b, red, tid);
    }
    __syncthreads();
    for (int idx = tid; idx < NCHAN * 512; idx += 128) {
        int c = idx >> 9, f = idx & 511;
        float acc = b1[f];
        const float* wr = w1 + (size_t)f * D128;
        for (int e = 0; e < D128; ++e) acc += sZ[c][e] * wr[e];
        sH1[idx] = gelu_f(acc);
    }
    __syncthreads();
    for (int c = 0; c < NCHAN; ++c) {
        float ff = b2[d];
        const float* wr = w2 + (size_t)d * 512;
        const float* hh = &sH1[c * 512];
        for (int f = 0; f < 512; ++f) ff += hh[f] * wr[f];
        sZ2[c][d] = layernorm_val(sZ[c][d] + ff, ln2_g, ln2_b, red, tid);
    }
    __syncthreads();
    sZm[d] = (sZ2[0][d] + sZ2[1][d] + sZ2[2][d]) * (1.0f / 3.0f);
    __syncthreads();
    float hv = h_b1[d];
    {
        const float* wr = h_w1 + (size_t)d * D128;
        for (int e = 0; e < D128; ++e) hv += sZm[e] * wr[e];
    }
    hv = gelu_f(hv);
    float tot = block_reduce_sum(hv * h_w2[d], red, tid);
    if (tid == 0) outp[b] = tot + h_b2[0];
}

// ---------------------------------------------------------------------------
// Host launcher.  Params arrive flattened in setup_inputs() dict order:
// d_in[0]=eeg_bands, d_in[1]=stage, d_in[2+i] = i-th params entry.
// Workspace peak: y1(236MB) + y2(236MB) + ~1MB small buffers; feat reuses y1.
// ---------------------------------------------------------------------------
extern "C" void kernel_launch(void* const* d_in, const int* in_sizes, int n_in,
                              void* d_out, int out_size, void* d_ws, size_t ws_size,
                              hipStream_t stream) {
    (void)in_sizes; (void)n_in; (void)out_size; (void)ws_size;
    const float* eeg = (const float*)d_in[0];
    const int* stage = (const int*)d_in[1];
    auto pf = [&](int i) { return (const float*)d_in[2 + i]; };
    // param indices (dict insertion order):
    // 0 c1_w 1-4 bn1 g/b/m/v  5 c2_w 6-9 bn2  10 c3_w 11-14 bn3
    // 15 pool_w 16 pool_b 17 band_emb 18 chan_emb 19 stage_emb 20 stage_token
    // 21 a0_in_w 22 a0_in_b 23 a0_out_w 24 a0_out_b
    // 25 a1_in_w 26 a1_in_b 27 a1_out_w 28 a1_out_b
    // 29 ln0_g 30 ln0_b 31 ln1_g 32 ln1_b 33 ln2_g 34 ln2_b
    // 35-38 ff0 w1/b1/w2/b2  39-42 ff1 w1/b1/w2/b2  43-46 h_w1/h_b1/h_w2/h_b2

    char* base = (char*)d_ws;
    size_t off = 0;
    auto alloc = [&](size_t bytes) -> void* {
        void* p = base + off;
        off += (bytes + 255) & ~(size_t)255;
        return p;
    };
    float*  W1f = (float*)alloc(C1 * 9 * sizeof(float));
    float*  B1  = (float*)alloc(C1 * sizeof(float));
    half_t* W2h = (half_t*)alloc((size_t)C2 * 224 * sizeof(half_t));
    float*  B2  = (float*)alloc(C2 * sizeof(float));
    half_t* W3h = (half_t*)alloc((size_t)C3 * 320 * sizeof(half_t));
    float*  B3  = (float*)alloc(C3 * sizeof(float));
    float*  tok = (float*)alloc((size_t)NSEQ * D128 * sizeof(float));
    float*  cs  = (float*)alloc((size_t)BATCH * NCHAN * D128 * sizeof(float));
    half_t* y1  = (half_t*)alloc((size_t)NSEQ * C1 * T1 * sizeof(half_t));
    half_t* y2  = (half_t*)alloc((size_t)NSEQ * C2 * T2 * sizeof(half_t));
    half_t* feat = y1;   // y1 is dead once conv2 finishes; reuse for conv3 output

    prep_kernel<<<160, 256, 0, stream>>>(
        pf(0), pf(1), pf(2), pf(3), pf(4),
        pf(5), pf(6), pf(7), pf(8), pf(9),
        pf(10), pf(11), pf(12), pf(13), pf(14),
        W1f, B1, W2h, B2, W3h, B3);

    conv1_kernel<<<NSEQ, 256, 0, stream>>>(eeg, W1f, B1, y1);
    conv2_wmma_kernel<<<dim3(15, NSEQ), 128, 0, stream>>>(y1, W2h, B2, y2);
    conv3_wmma_kernel<<<dim3(15, NSEQ), 256, 0, stream>>>(y2, W3h, B3, feat);
    attnpool_kernel<<<NSEQ, 128, 0, stream>>>(feat, pf(15), pf(16), pf(17), pf(18), tok);
    attn0_kernel<<<BATCH * NCHAN, 128, 0, stream>>>(
        stage, tok, pf(20), pf(19),
        pf(21), pf(22), pf(23), pf(24),
        pf(29), pf(30),
        pf(35), pf(36), pf(37), pf(38), cs);
    attn1_kernel<<<BATCH, 128, 0, stream>>>(
        cs,
        pf(25), pf(26), pf(27), pf(28),
        pf(31), pf(32), pf(33), pf(34),
        pf(39), pf(40), pf(41), pf(42),
        pf(43), pf(44), pf(45), pf(46),
        (float*)d_out);
}
